// NPair_loss_43035572306063
// MI455X (gfx1250) — matrix operand order repaired and checked by reading, not compiled, not run on hardware
//
#include <hip/hip_runtime.h>
#include <hip/hip_bf16.h>
#include <math.h>

// ---------------------------------------------------------------------------
// NPair loss, fused for MI455X (gfx1250, wave32, WMMA 16x16x32 bf16)
//   D = E E^T (rows 0..N-1 vs all 8192), s[m] = sum_n mask[m][n]*exp(D-ap[m])
//   out = mean(log1p(s))
// B tiles staged to LDS with gfx1250 async copies (double-buffered).
// ---------------------------------------------------------------------------

typedef __attribute__((ext_vector_type(16))) __bf16 v16bf;
typedef __attribute__((ext_vector_type(8)))  __bf16 v8bf;
typedef __attribute__((ext_vector_type(4)))  __bf16 v4bf;
typedef __attribute__((ext_vector_type(8)))  float  v8f;

#define BTOT  8192
#define NHALF 4096
#define KDIM  256

#define CHUNK_COLS 32            // columns staged per chunk (2 n-tiles)
#define LDS_STRIDE 264           // bf16 per staged row: 256 + 8 pad (528B = 132 dw)
#define NCHUNKS    16            // 512 cols per block / 32

// ------------------------- kernel 1: f32 -> bf16 ---------------------------
__global__ void cvt_bf16_kernel(const float* __restrict__ e,
                                __bf16* __restrict__ ebf) {
    int idx = (blockIdx.x * blockDim.x + threadIdx.x) * 4;
    float4 f = *(const float4*)(e + idx);
    v4bf o;
    o.x = (__bf16)f.x; o.y = (__bf16)f.y; o.z = (__bf16)f.z; o.w = (__bf16)f.w;
    *(v4bf*)(ebf + idx) = o;
}

// ------------------------- kernel 2: anchor-positive dots ------------------
__global__ void ap_kernel(const float* __restrict__ e, float* __restrict__ ap) {
    int i = blockIdx.x * blockDim.x + threadIdx.x;   // 0..NHALF-1
    const float4* a = (const float4*)(e + (size_t)i * KDIM);
    const float4* b = (const float4*)(e + (size_t)(i + NHALF) * KDIM);
    float acc = 0.f;
#pragma unroll 8
    for (int k = 0; k < KDIM / 4; ++k) {
        float4 x = a[k], y = b[k];
        acc += x.x * y.x + x.y * y.y + x.z * y.z + x.w * y.w;
    }
    ap[i] = acc;
}

// ------------------------- kernel 3: zero row sums -------------------------
__global__ void zero_kernel(float* __restrict__ s) {
    int i = blockIdx.x * blockDim.x + threadIdx.x;
    s[i] = 0.f;
}

// ------------------------- kernel 4: fused WMMA GEMM + epilogue ------------
// Grid: x = 16 column slabs (512 cols each), y = 32 row slabs (128 rows each)
// Block: 256 threads = 8 waves; wave w owns rows m0 = by*128 + w*16 .. +15
__global__ __launch_bounds__(256)
void npair_main_kernel(const __bf16* __restrict__ ebf,
                       const float* __restrict__ ap,
                       const unsigned char* __restrict__ negmask,
                       float* __restrict__ s) {
    // Double-buffered B slab: 32 rows x 264 bf16 x 2 buffers = 33,792 B
    __shared__ __bf16 sB[2][CHUNK_COLS * LDS_STRIDE];

    const int tid  = threadIdx.x;
    const int lane = tid & 31;
    const int wave = tid >> 5;
    const int lo   = lane & 15;       // column / row-within-tile index
    const int hi   = lane >> 4;       // K sub-block / row-half select
    const int m0   = blockIdx.y * 128 + wave * 16;
    const int nb0  = blockIdx.x * 512;

    // ---- Load A (16 rows x K=256) resident in VGPRs, WMMA A layout:
    //      lane(lo,hi): VGPR0-3 <- K = k0+8*hi .. +7 ; VGPR4-7 <- K+16
    v16bf a[8];
    {
        const __bf16* arow = ebf + (size_t)(m0 + lo) * KDIM;
#pragma unroll
        for (int st = 0; st < 8; ++st) {
            const int k0 = st * 32 + hi * 8;
            union { v16bf v; v8bf h[2]; } u;
            u.h[0] = *(const v8bf*)(arow + k0);
            u.h[1] = *(const v8bf*)(arow + k0 + 16);
            a[st] = u.v;
        }
    }

    // ap for the 8 rows this lane's accumulators map to (row = v + 8*hi)
    float apr[8];
#pragma unroll
    for (int v = 0; v < 8; ++v) apr[v] = ap[m0 + v + 8 * hi];

    float rowsum[8] = {0.f, 0.f, 0.f, 0.f, 0.f, 0.f, 0.f, 0.f};

    // ---- async staging: chunk = 32 E-rows (B columns) x 512B
    //      1024 x b128 per chunk, 4 per thread -> 4 async ops per wave
    auto stage = [&](int c, int buf) {
        const char* gbase =
            (const char*)(ebf + (size_t)(nb0 + c * CHUNK_COLS) * KDIM);
        unsigned int lbase = (unsigned int)(size_t)(&sB[buf][0]);
#pragma unroll
        for (int j = 0; j < 4; ++j) {
            int q   = tid + j * 256;           // 0..1023
            int row = q >> 5;                  // 0..31
            int off = (q & 31) * 16;           // 0..496
            unsigned long long gp =
                (unsigned long long)(size_t)(gbase + row * 512 + off);
            unsigned int lp = lbase + (unsigned int)(row * (LDS_STRIDE * 2) + off);
            asm volatile("global_load_async_to_lds_b128 %0, %1, off"
                         :: "v"(lp), "v"(gp) : "memory");
        }
    };

    stage(0, 0);
    for (int c = 0; c < NCHUNKS; ++c) {
        if (c + 1 < NCHUNKS) {
            stage(c + 1, (c + 1) & 1);
            asm volatile("s_wait_asynccnt 0x4" ::: "memory");   // chunk c done
        } else {
            asm volatile("s_wait_asynccnt 0x0" ::: "memory");
        }
        __syncthreads();                       // everyone's chunk-c data in LDS

        const __bf16* bufp = &sB[c & 1][0];
#pragma unroll
        for (int nl = 0; nl < 2; ++nl) {
            const int n0 = nb0 + c * CHUNK_COLS + nl * 16;
            // B layout: lane(lo,hi) holds B column n0+lo, K = 8*hi..+7 and 16+8*hi..+7
            const __bf16* bbase = bufp + (nl * 16 + lo) * LDS_STRIDE + hi * 8;

            // prefetch mask bytes so their latency hides under the WMMAs
            unsigned char mk[8];
            const unsigned char* mcol = negmask + (n0 + lo);
#pragma unroll
            for (int v = 0; v < 8; ++v)
                mk[v] = mcol[(size_t)(m0 + v + 8 * hi) * BTOT];

            v8f cacc = {};
#pragma unroll
            for (int st = 0; st < 8; ++st) {
                union { v16bf v; v8bf h[2]; } u;
                u.h[0] = *(const v8bf*)(bbase + st * 32);
                u.h[1] = *(const v8bf*)(bbase + st * 32 + 16);
                cacc = __builtin_amdgcn_wmma_f32_16x16x32_bf16(
                        /*neg_a=*/false, a[st], /*neg_b=*/false, u.v,
                        /*c_mod=*/(short)0, cacc,
                        /*reuse_a=*/false, /*reuse_b=*/false);
            }

#pragma unroll
            for (int v = 0; v < 8; ++v) {
                float ex = __expf(cacc[v] - apr[v]);
                rowsum[v] += mk[v] ? ex : 0.f;
            }
        }
        __syncthreads();                       // buf[c&1] free for chunk c+2
    }

    // ---- Reduce the 16-lane column groups, one atomic per row
#pragma unroll
    for (int v = 0; v < 8; ++v) {
        float r = rowsum[v];
#pragma unroll
        for (int off = 1; off < 16; off <<= 1)
            r += __shfl_xor(r, off, 16);
        if (lo == 0)
            atomicAdd(&s[m0 + v + 8 * hi], r);
    }
}

// ------------------------- kernel 5: mean(log1p(s)) ------------------------
__global__ void finish_kernel(const float* __restrict__ s, float* __restrict__ out) {
    __shared__ float red[256];
    float acc = 0.f;
    for (int i = threadIdx.x; i < NHALF; i += 256)
        acc += log1pf(s[i]);
    red[threadIdx.x] = acc;
    __syncthreads();
    for (int st = 128; st > 0; st >>= 1) {
        if (threadIdx.x < st) red[threadIdx.x] += red[threadIdx.x + st];
        __syncthreads();
    }
    if (threadIdx.x == 0) out[0] = red[0] * (1.0f / NHALF);
}

// ---------------------------------------------------------------------------
extern "C" void kernel_launch(void* const* d_in, const int* in_sizes, int n_in,
                              void* d_out, int out_size, void* d_ws, size_t ws_size,
                              hipStream_t stream) {
    const float*         emb     = (const float*)d_in[0];
    // d_in[1] = pos_mask (unused by the reference computation)
    const unsigned char* negmask = (const unsigned char*)d_in[2];
    float* out = (float*)d_out;

    // Workspace layout
    char* ws = (char*)d_ws;
    __bf16* ebf = (__bf16*)ws;                                  // 4 MiB
    float*  ap  = (float*)(ws + (size_t)BTOT * KDIM * 2);       // 16 KiB
    float*  s   = ap + NHALF;                                   // 16 KiB

    // 1) convert embeddings to bf16 (8192*256 / (256*4) = 2048 blocks)
    cvt_bf16_kernel<<<dim3(2048), dim3(256), 0, stream>>>(emb, ebf);
    // 2) anchor-positive dot products (fp32 for accuracy)
    ap_kernel<<<dim3(NHALF / 256), dim3(256), 0, stream>>>(emb, ap);
    // 3) zero the row-sum accumulators
    zero_kernel<<<dim3(NHALF / 256), dim3(256), 0, stream>>>(s);
    // 4) fused WMMA GEMM + masked-exp row reduction (B via async-LDS)
    npair_main_kernel<<<dim3(16, 32), dim3(256), 0, stream>>>(ebf, ap, negmask, s);
    // 5) mean(log1p(s)) -> scalar
    finish_kernel<<<dim3(1), dim3(256), 0, stream>>>(s, out);
}
